// MultiHeadAttention_48799418417161
// MI455X (gfx1250) — compile-verified, matrix operand order checked by reading, not run on hardware
//
#include <hip/hip_runtime.h>
#include <hip/hip_bf16.h>

// ---------------------------------------------------------------------------
// MultiHeadAttention for MI455X (gfx1250, wave32, WMMA).
// Precision: bf16 WMMA with f32 accumulate (kernel is HBM-store bound on the
// 537MB attention-weight tensor, so bf16 matrix math is free performance-wise).
// ---------------------------------------------------------------------------

#define NB 8
#define SEQ 1024
#define DM 1024
#define NH 16
#define HD 64
#define NEGV (-1e9f)

typedef __attribute__((ext_vector_type(16))) __bf16 v16bf;
typedef __attribute__((ext_vector_type(2)))  __bf16 v2bf;
typedef __attribute__((ext_vector_type(8)))  float  v8f;

static __device__ __forceinline__ unsigned short f32_to_bf16(float f) {
  unsigned int u = __builtin_bit_cast(unsigned int, f);
  u += 0x7FFFu + ((u >> 16) & 1u);           // round-to-nearest-even
  return (unsigned short)(u >> 16);
}
static __device__ __forceinline__ __bf16 bc16(unsigned int u) {
  return __builtin_bit_cast(__bf16, (unsigned short)(u & 0xFFFFu));
}

// Convert a pair of f32 to packed bf16 (one dword).
static __device__ __forceinline__ unsigned int pk_bf16(float a, float b) {
#if __has_builtin(__builtin_amdgcn_cvt_pk_bf16_f32)
  v2bf p = __builtin_amdgcn_cvt_pk_bf16_f32(a, b);
  return __builtin_bit_cast(unsigned int, p);
#else
  return (unsigned int)f32_to_bf16(a) | ((unsigned int)f32_to_bf16(b) << 16);
#endif
}

// Convert 8 f32 -> 8 bf16 and store as 4 dwords at d (4-byte aligned).
static __device__ __forceinline__ void stash8(unsigned short* d, float4 a, float4 b) {
  unsigned int* d32 = (unsigned int*)d;
  d32[0] = pk_bf16(a.x, a.y);
  d32[1] = pk_bf16(a.z, a.w);
  d32[2] = pk_bf16(b.x, b.y);
  d32[3] = pk_bf16(b.z, b.w);
}

// Build one 16x32 bf16 operand fragment (8 VGPRs / lane) from a 32-element
// contiguous bf16 run `p` (the lane's M-row for A, or N-column vector for B).
// Lanes 0-15 take K in {0..7,16..23}; lanes 16-31 take K in {8..15,24..31}.
static __device__ __forceinline__ v16bf build_frag(const unsigned short* p, int half) {
  v16bf f;
  const int s0 = half ? 8 : 0;
#pragma unroll
  for (int j = 0; j < 4; ++j) {
    unsigned int u0 = *(const unsigned int*)(p + s0 + 2 * j);
    unsigned int u1 = *(const unsigned int*)(p + s0 + 16 + 2 * j);
    f[2 * j]     = bc16(u0);        f[2 * j + 1]     = bc16(u0 >> 16);
    f[8 + 2 * j] = bc16(u1);        f[8 + 2 * j + 1] = bc16(u1 >> 16);
  }
  return f;
}

static __device__ __forceinline__ v8f wmma_bf16(v16bf a, v16bf b, v8f c) {
  return __builtin_amdgcn_wmma_f32_16x16x32_bf16(false, a, false, b, (short)0, c,
                                                 false, false);
}

// ---------------------------------------------------------------------------
// Kernel 1: GEMM  out[m,n] = A[m,:1024] . W[n,:1024] + bias[n]
//   mode 0: bf16 out, layout [N,H,S,D]      (Q / K projections)
//   mode 1: bf16 out, layout [N,H,D,S]      (V projection, pre-transposed)
//   mode 2: f32  out, layout [M, 1024]      (output projection)
// Block = 256 threads (8 waves). Block tile 64x64, wave tile 16x32, K-step 32.
// Software-pipelined: register prefetch of tile i+1 overlaps WMMAs of tile i;
// ping-pong LDS buffers -> one barrier per K-step, loadcnt wait lands after
// the WMMAs instead of before them.
// ---------------------------------------------------------------------------
__global__ void __launch_bounds__(256)
proj_gemm(const float* __restrict__ A, const float* __restrict__ W,
          const float* __restrict__ bias, unsigned short* __restrict__ obf,
          float* __restrict__ of32, int mode) {
  __shared__ unsigned short As[2][64 * 32];
  __shared__ unsigned short Ws[2][64 * 32];

  const int tid = threadIdx.x;
  const int wave = tid >> 5, lane = tid & 31;
  const int half = lane >> 4, lane16 = lane & 15;
  const int M0 = blockIdx.x * 64, N0 = blockIdx.y * 64;
  const int mw = (wave & 3) * 16, nw = (wave >> 2) * 32;

  const int srow = tid >> 2;            // staging: 64 rows x 32 cols, 8/thread
  const int scol = (tid & 3) * 8;
  const int sidx = srow * 32 + scol;

  const float* pA = A + (size_t)(M0 + srow) * DM + scol;   // this thread's rows
  const float* pW = W + (size_t)(N0 + srow) * DM + scol;

  // ---- prologue: stage K-tile 0 -------------------------------------------
  {
    float4 a0 = *(const float4*)(pA);
    float4 a1 = *(const float4*)(pA + 4);
    float4 w0 = *(const float4*)(pW);
    float4 w1 = *(const float4*)(pW + 4);
    stash8(&As[0][sidx], a0, a1);
    stash8(&Ws[0][sidx], w0, w1);
  }
  __syncthreads();

  v8f acc0 = {}, acc1 = {};

  for (int it = 0; it < DM / 32; ++it) {
    const int cur = it & 1;
    const bool more = (it + 1 < DM / 32);

    // prefetch next K-tile into registers (no wait until the stash below)
    float4 na0, na1, nw0, nw1;
    if (more) {
      const float* qa = pA + (it + 1) * 32;
      const float* qw = pW + (it + 1) * 32;
      na0 = *(const float4*)(qa);
      na1 = *(const float4*)(qa + 4);
      nw0 = *(const float4*)(qw);
      nw1 = *(const float4*)(qw + 4);
      if (it + 2 < DM / 32) {   // gfx1250 global_prefetch_b8, distance 2 tiles
        __builtin_prefetch((const char*)(qa + 32), 0, 3);
        __builtin_prefetch((const char*)(qw + 32), 0, 3);
      }
    }

    // compute on current buffer while the loads are in flight
    v16bf a  = build_frag(&As[cur][(mw + lane16) * 32], half);
    v16bf b0 = build_frag(&Ws[cur][(nw + lane16) * 32], half);
    v16bf b1 = build_frag(&Ws[cur][(nw + 16 + lane16) * 32], half);
    acc0 = wmma_bf16(a, b0, acc0);
    acc1 = wmma_bf16(a, b1, acc1);

    if (more) {
      stash8(&As[cur ^ 1][sidx], na0, na1);
      stash8(&Ws[cur ^ 1][sidx], nw0, nw1);
      __syncthreads();
    }
  }

  // Store: C layout -> element r of lane: M = r + 8*half, N = lane16.
#pragma unroll
  for (int tile = 0; tile < 2; ++tile) {
    v8f acc = tile ? acc1 : acc0;
#pragma unroll
    for (int r = 0; r < 8; ++r) {
      const int m = M0 + mw + r + half * 8;
      const int nc = N0 + nw + tile * 16 + lane16;
      const float v = acc[r] + bias[nc];
      if (mode == 2) {
        of32[(size_t)m * DM + nc] = v;
      } else {
        const int n = m >> 10, s = m & (SEQ - 1);
        const int h = nc >> 6, d = nc & (HD - 1);
        size_t off = (mode == 0)
                         ? ((((size_t)n * NH + h) * SEQ + s) * HD + d)
                         : ((((size_t)n * NH + h) * HD + d) * SEQ + s);
        obf[off] = f32_to_bf16(v);
      }
    }
  }
}

// ---------------------------------------------------------------------------
// Kernel 2: attention for one (n, h, 16-query tile).
// Phase 1: scores (8 waves x 16q x 128k, WMMA QK^T, scale + masks) -> LDS f32.
// Phase 2: row softmax in LDS, stream w to HBM, keep bf16 probs in LDS.
// Phase 3: x_tile = P @ V via WMMA (waves 0..3, one 16-wide d tile each).
// Dynamic LDS: 16*1024*4 (scores) + 16*1024*2 (probs) + reductions ~= 97KB.
// ---------------------------------------------------------------------------
__global__ void __launch_bounds__(256)
attn(const unsigned short* __restrict__ Qp, const unsigned short* __restrict__ Kp,
     const unsigned short* __restrict__ VpT, const int* __restrict__ padmask,
     float* __restrict__ out_w, float* __restrict__ xbuf) {
  extern __shared__ char smem[];
  float*          sc     = (float*)smem;                              // [16][1024]
  unsigned short* probs  = (unsigned short*)(smem + 65536);           // [16][1024]
  float*          red    = (float*)(smem + 65536 + 32768);            // [16][16]
  float*          rowmax = red + 256;                                 // [16]
  float*          rowsum = rowmax + 16;                               // [16]

  const int b = blockIdx.x;
  const int qt = b & 63, h = (b >> 6) & (NH - 1), n = b >> 10;
  const int qbase = qt * 16;

  const int tid = threadIdx.x;
  const int wave = tid >> 5, lane = tid & 31;
  const int half = lane >> 4, lane16 = lane & 15;

  const size_t headQ = ((size_t)n * NH + h) * SEQ;   // rows of Qp/Kp for head
  const size_t headV = ((size_t)n * NH + h) * HD;    // rows of VpT for head

  // ---- Phase 1: scores -----------------------------------------------------
  {
    const unsigned short* qrow = Qp + (headQ + qbase + lane16) * HD;
    v16bf aq0 = build_frag(qrow, half);
    v16bf aq1 = build_frag(qrow + 32, half);

    const int kbase = wave * 128;
#pragma unroll
    for (int kt = 0; kt < 8; ++kt) {
      const int k0 = kbase + kt * 16;
      const unsigned short* krow = Kp + (headQ + k0 + lane16) * HD;
      v16bf b0 = build_frag(krow, half);
      v16bf b1 = build_frag(krow + 32, half);
      v8f c = {};
      c = wmma_bf16(aq0, b0, c);
      c = wmma_bf16(aq1, b1, c);
#pragma unroll
      for (int r = 0; r < 8; ++r) {
        const int M = r + half * 8;
        const int qg = qbase + M;
        const int kg = k0 + lane16;
        float s = c[r] * 0.03125f;                 // 1/sqrt(d_model=1024)
        if (kg > qg) s = NEGV;                     // causal (strict upper tri)
        if (padmask[n * SEQ + kg] == 1) s = NEGV;  // key padding
        sc[M * SEQ + kg] = s;
      }
    }
  }
  __syncthreads();

  // ---- Phase 2: softmax + emit w ------------------------------------------
  {
    const int row = tid >> 4;      // 16 threads per row
    const int c0 = tid & 15;
    float mx = -3.4e38f;
#pragma unroll 8
    for (int i = 0; i < 64; ++i) mx = fmaxf(mx, sc[row * SEQ + c0 + 16 * i]);
    red[row * 16 + c0] = mx;
    __syncthreads();
    if (c0 == 0) {
      float m = -3.4e38f;
#pragma unroll
      for (int j = 0; j < 16; ++j) m = fmaxf(m, red[row * 16 + j]);
      rowmax[row] = m;
    }
    __syncthreads();
    const float m = rowmax[row];
    float sum = 0.f;
#pragma unroll 8
    for (int i = 0; i < 64; ++i) {
      const int idx = row * SEQ + c0 + 16 * i;
      const float e = __expf(sc[idx] - m);
      sc[idx] = e;
      sum += e;
    }
    red[row * 16 + c0] = sum;
    __syncthreads();
    if (c0 == 0) {
      float t = 0.f;
#pragma unroll
      for (int j = 0; j < 16; ++j) t += red[row * 16 + j];
      rowsum[row] = t;
    }
    __syncthreads();
    const float inv = 1.0f / rowsum[row];
    float* wrow = out_w + ((((size_t)n * NH + h) * SEQ) + qbase + row) * SEQ;
#pragma unroll 8
    for (int i = 0; i < 64; ++i) {
      const int col = c0 + 16 * i;
      const float p = sc[row * SEQ + col] * inv;
      wrow[col] = p;                       // 537MB stream: the HBM bottleneck
      probs[row * SEQ + col] = f32_to_bf16(p);
    }
  }
  __syncthreads();

  // ---- Phase 3: x = P @ V (waves 0..3, d-tile = wave*16) -------------------
  if (wave < 4) {
    const int d0 = wave * 16;
    v8f acc = {};
    const unsigned short* arow = probs + lane16 * SEQ;
    const unsigned short* vrow = VpT + (headV + d0 + lane16) * SEQ;
#pragma unroll 4
    for (int kt = 0; kt < 32; ++kt) {
      const int k0 = kt * 32;
      v16bf a = build_frag(arow + k0, half);
      v16bf bb = build_frag(vrow + k0, half);
      acc = wmma_bf16(a, bb, acc);
    }
#pragma unroll
    for (int r = 0; r < 8; ++r) {
      const int s = qbase + r + half * 8;
      const int col = h * HD + d0 + lane16;
      xbuf[((size_t)n * SEQ + s) * DM + col] = acc[r];
    }
  }
}

// ---------------------------------------------------------------------------
extern "C" void kernel_launch(void* const* d_in, const int* in_sizes, int n_in,
                              void* d_out, int out_size, void* d_ws, size_t ws_size,
                              hipStream_t stream) {
  const float* q  = (const float*)d_in[0];
  const float* k  = (const float*)d_in[1];
  const float* v  = (const float*)d_in[2];
  // d_in[3] = causal_mask (recomputed analytically in-kernel)
  const int*   pm = (const int*)d_in[4];
  const float* Wq = (const float*)d_in[5];  const float* bq = (const float*)d_in[6];
  const float* Wk = (const float*)d_in[7];  const float* bk = (const float*)d_in[8];
  const float* Wv = (const float*)d_in[9];  const float* bv = (const float*)d_in[10];
  const float* Wo = (const float*)d_in[11]; const float* bo = (const float*)d_in[12];

  float* out_x = (float*)d_out;                         // [8,1024,1024]
  float* out_w = out_x + (size_t)NB * SEQ * DM;         // [8,16,1024,1024]

  // workspace carve-up (needs ~84MB)
  const size_t BF = (size_t)NB * NH * SEQ * HD;         // 8.39M elems
  char* ws = (char*)d_ws;
  unsigned short* Qp  = (unsigned short*)ws;
  unsigned short* Kp  = (unsigned short*)(ws + 2 * BF);
  unsigned short* VpT = (unsigned short*)(ws + 4 * BF);
  float*          xb  = (float*)(ws + 6 * BF);

  dim3 pg(NB * SEQ / 64, DM / 64);                      // (128, 16)
  proj_gemm<<<pg, 256, 0, stream>>>(q, Wq, bq, Qp, nullptr, 0);
  proj_gemm<<<pg, 256, 0, stream>>>(k, Wk, bk, Kp, nullptr, 0);
  proj_gemm<<<pg, 256, 0, stream>>>(v, Wv, bv, VpT, nullptr, 1);

  const size_t smem = 65536 + 32768 + 256 * 4 + 16 * 4 + 16 * 4;  // ~97KB
  attn<<<NB * NH * (SEQ / 16), 256, smem, stream>>>(Qp, Kp, VpT, pm, out_w, xb);

  proj_gemm<<<pg, 256, 0, stream>>>(xb, Wo, bo, nullptr, out_x, 2);
}
